// emb_att_Layers_34479997452528
// MI455X (gfx1250) — matrix operand order repaired and checked by reading, not compiled, not run on hardware
//
#include <hip/hip_runtime.h>
#include <hip/hip_bf16.h>
#include <stdint.h>

// ---------------------------------------------------------------------------
// Problem constants (match reference)
// ---------------------------------------------------------------------------
#define NN      50000
#define EE      1000000
#define RR      16
#define DD      128
#define HH      4
#define HIDC    128
#define LL      32
#define SS      4

typedef __attribute__((ext_vector_type(16))) __bf16 bf16x16;
typedef __attribute__((ext_vector_type(8)))  float  f32x8;
typedef __attribute__((ext_vector_type(4)))  unsigned int u32x4;
typedef __attribute__((ext_vector_type(8)))  int i32x8;
typedef __attribute__((ext_vector_type(4)))  int i32x4;

// ---------------------------------------------------------------------------
// WMMA fragment helpers (CDNA5 wave32, V_WMMA_F32_16X16X32_BF16)
// A (16x32 bf16): lane m in [0,15] holds row m with K={0..7,16..23} (2/VGPR),
//                 lane m+16 holds row m with K={8..15,24..31}.
// B (32x16 bf16): lane n in [0,15] holds col n with K={0..7,16..23},
//                 lane n+16 holds col n with K={8..15,24..31}.
// B matrices are stored PRE-SWIZZLED: per 32x16 block, lane-major, 16 contig
// bf16 per lane -> fragment load is a single 32-byte vector load.
// C/D (16x16 f32): lane n holds col n; VGPR i holds row i + 8*(lane>=16).
// ---------------------------------------------------------------------------
__device__ __forceinline__ bf16x16 load_a_frag(const __bf16* A, int lda, int lane) {
    int m = lane & 15, half = lane >> 4;
    union { bf16x16 v; unsigned int u[8]; } r;
    const __bf16* row = A + (size_t)m * lda + half * 8;
#pragma unroll
    for (int j = 0; j < 4; ++j) r.u[j]     = *(const unsigned int*)(row + 2 * j);
#pragma unroll
    for (int j = 0; j < 4; ++j) r.u[4 + j] = *(const unsigned int*)(row + 16 + 2 * j);
    return r.v;
}

// Bblk points at base of a pre-swizzled 32x16 block (512 bf16 = 1 KB)
__device__ __forceinline__ bf16x16 load_b_frag_sw(const __bf16* Bblk, int lane) {
    return *(const bf16x16*)(Bblk + lane * 16);
}

__device__ __forceinline__ f32x8 wmma_bf16(bf16x16 a, bf16x16 b, f32x8 c) {
    return __builtin_amdgcn_wmma_f32_16x16x32_bf16(false, a, false, b, (short)0, c,
                                                   false, false);
}

// swizzled index of logical element B[k][n] in a K x ncols matrix
__device__ __forceinline__ size_t bswz_index(int k, int n, int ncols) {
    int kblk = k >> 5, nblk = n >> 4;
    int kin = k & 31, nin = n & 15;
    int half = (kin >> 3) & 1, j = (kin & 7) | ((kin >> 4) << 3);
    int lane = nin + (half << 4);
    return ((size_t)(kblk * (ncols >> 4) + nblk) * 32 + lane) * 16 + j;
}

// ---------------------------------------------------------------------------
// TDM: 1-D tensor_load_to_lds (ndwords DWORDs, global -> LDS), wave-scalar.
// D# group0: count=1, lds_addr, 57-bit global addr, type=2.
// D# group1: data_size=4B, tensor_dim0=tile_dim0=ndwords, stride=ndwords.
// clang-23 toolchain: 6-arg builtin (g0, g1, g2, g3, g4, cpol).
// ---------------------------------------------------------------------------
__device__ __forceinline__ void tdm_load_1d(unsigned lds_off, const void* gptr,
                                            unsigned ndwords) {
    unsigned long long ga = (unsigned long long)(uintptr_t)gptr;
    u32x4 g0;
    g0[0] = 1u;                                     // count=1, user mode
    g0[1] = lds_off;                                // lds_addr (bytes)
    g0[2] = (unsigned)(ga & 0xffffffffu);           // global_addr[31:0]
    g0[3] = (unsigned)((ga >> 32) & 0x01ffffffu)    // global_addr[56:32]
          | 0x80000000u;                            // type=2 (image)
    i32x8 g1;
    g1[0] = (int)(2u << 16);                        // data_size=4B
    g1[1] = (int)((ndwords & 0xffffu) << 16);       // tensor_dim0[15:0]
    g1[2] = (int)(((ndwords >> 16) & 0xffffu)       // tensor_dim0[31:16]
          | (1u << 16));                            // tensor_dim1[15:0]=1
    g1[3] = (int)((ndwords & 0xffffu) << 16);       // tile_dim0
    g1[4] = 0;                                      // tile_dim1=0, tile_dim2=0
    g1[5] = (int)ndwords;                           // tensor_dim0_stride[31:0]
    g1[6] = (int)((ndwords >> 16) & 0xffffu);       // tensor_dim0_stride[47:32]
    g1[7] = 0;
    i32x4 gz4 = {0, 0, 0, 0};
    i32x8 gz8 = {0, 0, 0, 0, 0, 0, 0, 0};
    __builtin_amdgcn_tensor_load_to_lds(g0, g1, gz4, gz4, gz8, 0);
}

// ---------------------------------------------------------------------------
// Kernel 0: convert weights to bf16, pre-swizzled into WMMA B-fragment order
// ---------------------------------------------------------------------------
__global__ void convert_weights_kernel(const float* wi, const float* wo,
                                       const float* w1, const float* r1,
                                       const float* w2, const float* r2,
                                       __bf16* wiS, __bf16* woS,
                                       __bf16* w1S, __bf16* r1S,
                                       __bf16* w2S, __bf16* r2S) {
    int gid = blockIdx.x * blockDim.x + threadIdx.x;
    int idx = gid;
    if (idx < 3 * DD * DD) {                 // in_proj_w [384,128]: B[k][n]=wi[n][k]
        int n = idx / DD, k = idx % DD;
        wiS[bswz_index(k, n, 3 * DD)] = (__bf16)wi[idx];
        return;
    }
    idx -= 3 * DD * DD;
    if (idx < DD * DD) {                     // out_proj_w: B[k][n]=wo[n][k]
        int n = idx / DD, k = idx % DD;
        woS[bswz_index(k, n, DD)] = (__bf16)wo[idx];
        return;
    }
    idx -= DD * DD;
    if (idx < RR * DD * HIDC) {              // w1[r][k][n]
        int r = idx / (DD * HIDC), rem = idx % (DD * HIDC);
        int k = rem / HIDC, n = rem % HIDC;
        w1S[(size_t)r * DD * HIDC + bswz_index(k, n, HIDC)] = (__bf16)w1[idx];
        return;
    }
    idx -= RR * DD * HIDC;
    if (idx < DD * HIDC) {                   // root1[k][n]
        int k = idx / HIDC, n = idx % HIDC;
        r1S[bswz_index(k, n, HIDC)] = (__bf16)r1[idx];
        return;
    }
    idx -= DD * HIDC;
    if (idx < RR * HIDC * LL) {              // w2[r][k][n]
        int r = idx / (HIDC * LL), rem = idx % (HIDC * LL);
        int k = rem / LL, n = rem % LL;
        w2S[(size_t)r * HIDC * LL + bswz_index(k, n, LL)] = (__bf16)w2[idx];
        return;
    }
    idx -= RR * HIDC * LL;
    if (idx < HIDC * LL) {                   // root2[k][n]
        int k = idx / LL, n = idx % LL;
        r2S[bswz_index(k, n, LL)] = (__bf16)r2[idx];
    }
}

// ---------------------------------------------------------------------------
// Kernel 1: fused MHA (seq len 4, output position 0) -> x0 bf16 [N,128]
// One block = 16 nodes, 256 threads (8 waves). B fragments straight from L2.
// ---------------------------------------------------------------------------
__global__ void mha_kernel(const float* __restrict__ emb,
                           const __bf16* __restrict__ wiS, const float* __restrict__ bi,
                           const __bf16* __restrict__ woS, const float* __restrict__ bo,
                           __bf16* __restrict__ x0, int n_nodes) {
    __shared__ __align__(32) __bf16 sEmb[64 * DD];   // 16 KB, reused as sO
    __shared__ __align__(32) __bf16 sQ[16 * DD];     // 4 KB
    __shared__ __align__(32) __bf16 sK[64 * DD];     // 16 KB
    __shared__ __align__(32) __bf16 sV[64 * DD];     // 16 KB

    const int tid = threadIdx.x;
    const int lane = tid & 31;
    const int wave = tid >> 5;
    const int node0 = blockIdx.x * 16;

    // ---- stage emb tile (fp32 -> bf16), rows = s*16+i ----
    for (int e = tid; e < 64 * DD; e += 256) {
        int row = e >> 7, col = e & 127;
        int s = row >> 4, i = row & 15;
        int n = node0 + i;
        float v = (n < n_nodes) ? emb[((size_t)s * n_nodes + n) * DD + col] : 0.0f;
        sEmb[e] = (__bf16)v;
    }
    __syncthreads();

    // ---- qkv GEMM: [64x128] @ wiS[128x384]; q only for s=0 rows ----
    for (int t = wave; t < 72; t += 8) {
        int kind, rt, ct;
        if (t < 8)       { kind = 0; rt = 0;             ct = t; }
        else if (t < 40) { kind = 1; rt = (t - 8) >> 3;  ct = (t - 8) & 7; }
        else             { kind = 2; rt = (t - 40) >> 3; ct = (t - 40) & 7; }
        int cb = kind * 8 + ct;  // global col block (of 24)
        f32x8 c = {0.f, 0.f, 0.f, 0.f, 0.f, 0.f, 0.f, 0.f};
#pragma unroll
        for (int kk = 0; kk < 4; ++kk) {
            bf16x16 a = load_a_frag(&sEmb[(rt * 16) * DD + kk * 32], DD, lane);
            bf16x16 b = load_b_frag_sw(wiS + (size_t)(kk * 24 + cb) * 512, lane);
            c = wmma_bf16(a, b, c);
        }
        int n = lane & 15, half = lane >> 4;
        int col = ct * 16 + n;
        float bias = bi[kind * DD + col];
#pragma unroll
        for (int i = 0; i < 8; ++i) {
            int m = i + half * 8;
            float v = c[i] + bias;
            int row = rt * 16 + m;
            if (kind == 0)      sQ[row * DD + col] = (__bf16)v;
            else if (kind == 1) sK[row * DD + col] = (__bf16)v;
            else                sV[row * DD + col] = (__bf16)v;
        }
    }
    __syncthreads();

    // ---- attention: thread = (node i, head h); seq len 4, head dim 32 ----
    __bf16* sO = sEmb;  // reuse
    if (tid < 64) {
        int i = tid & 15, h = tid >> 4;
        const float scale = 0.17677669529663687f;  // 1/sqrt(32)
        float q[32];
#pragma unroll
        for (int d = 0; d < 32; ++d) q[d] = (float)sQ[i * DD + h * 32 + d];
        float sc[4], mx = -1e30f;
#pragma unroll
        for (int t = 0; t < 4; ++t) {
            float s = 0.f;
#pragma unroll
            for (int d = 0; d < 32; ++d)
                s += q[d] * (float)sK[(t * 16 + i) * DD + h * 32 + d];
            sc[t] = s * scale;
            mx = fmaxf(mx, sc[t]);
        }
        float den = 0.f;
#pragma unroll
        for (int t = 0; t < 4; ++t) { sc[t] = __expf(sc[t] - mx); den += sc[t]; }
        float o[32];
#pragma unroll
        for (int d = 0; d < 32; ++d) o[d] = 0.f;
#pragma unroll
        for (int t = 0; t < 4; ++t)
#pragma unroll
            for (int d = 0; d < 32; ++d)
                o[d] += sc[t] * (float)sV[(t * 16 + i) * DD + h * 32 + d];
        float inv = 1.0f / den;
#pragma unroll
        for (int d = 0; d < 32; ++d) sO[i * DD + h * 32 + d] = (__bf16)(o[d] * inv);
    }
    __syncthreads();

    // ---- output projection: [16x128] @ woS[128x128] -> x0 ----
    {
        int ct = wave;  // 8 col tiles / 8 waves
        f32x8 c = {0.f, 0.f, 0.f, 0.f, 0.f, 0.f, 0.f, 0.f};
#pragma unroll
        for (int kk = 0; kk < 4; ++kk) {
            bf16x16 a = load_a_frag(&sO[kk * 32], DD, lane);
            bf16x16 b = load_b_frag_sw(woS + (size_t)(kk * 8 + ct) * 512, lane);
            c = wmma_bf16(a, b, c);
        }
        int n = lane & 15, half = lane >> 4;
        int col = ct * 16 + n;
        float bias = bo[col];
#pragma unroll
        for (int i = 0; i < 8; ++i) {
            int m = i + half * 8;
            int node = node0 + m;
            if (node < n_nodes) x0[(size_t)node * DD + col] = (__bf16)(c[i] + bias);
        }
    }
}

// ---------------------------------------------------------------------------
// Kernel 2: per-(relation,dst) edge counts
// ---------------------------------------------------------------------------
__global__ void count_kernel(const int* __restrict__ edge_index,
                             const int* __restrict__ edge_type,
                             float* __restrict__ cnt, int n_edges, int n_nodes) {
    int e = blockIdx.x * blockDim.x + threadIdx.x;
    if (e >= n_edges) return;
    int d = edge_index[n_edges + e];
    int r = edge_type[e];
    unsafeAtomicAdd(&cnt[(size_t)r * n_nodes + d], 1.0f);
}

// ---------------------------------------------------------------------------
// Kernel 3: edge aggregation: agg[r][dst][:] += x[src][:]
// One wave per edge; each lane owns 4 of the 128 features (coalesced 512B).
// ---------------------------------------------------------------------------
__global__ void aggregate_kernel(const int* __restrict__ edge_index,
                                 const int* __restrict__ edge_type,
                                 const __bf16* __restrict__ x,
                                 float* __restrict__ agg, int n_edges, int n_nodes) {
    long long gid = (long long)blockIdx.x * blockDim.x + threadIdx.x;
    int e = (int)(gid >> 5);
    int lane = (int)(gid & 31);
    if (e >= n_edges) return;
    int s = edge_index[e];
    int d = edge_index[n_edges + e];
    int r = edge_type[e];
    const __bf16* xs = x + (size_t)s * DD;
    float* ad = agg + ((size_t)r * n_nodes + d) * DD;
#pragma unroll
    for (int j = 0; j < 4; ++j) {
        int col = lane + j * 32;
        unsafeAtomicAdd(&ad[col], (float)xs[col]);
    }
}

// ---------------------------------------------------------------------------
// Kernel 4: RGCN GEMM: out = act( sum_r (agg_r/cnt_r) @ W_r + x @ root + b )
// Block = 64 nodes, 256 threads (8 waves). B staged via TDM tensor_load_to_lds.
// MODE 0: ReLU->bf16, MODE 1: sigmoid->f32
// ---------------------------------------------------------------------------
template <int OUTC, int MODE>
__global__ void rgcn_kernel(const float* __restrict__ agg, const float* __restrict__ cnt,
                            const __bf16* __restrict__ xin,
                            const __bf16* __restrict__ wrel, const __bf16* __restrict__ root,
                            const float* __restrict__ bias, void* __restrict__ out,
                            int n_nodes) {
    constexpr int NCT = OUTC / 16;          // col tiles
    constexpr int TPW = (4 * NCT) / 8;      // C tiles per wave
    __shared__ __align__(32) __bf16 sA[64 * DD];      // 16 KB (row-major)
    __shared__ __align__(32) __bf16 sB[DD * OUTC];    // swizzled blocks

    const int tid = threadIdx.x;
    const int lane = tid & 31;
    const int wave = tid >> 5;
    const int node0 = blockIdx.x * 64;
    const unsigned sB_off = (unsigned)(uintptr_t)(&sB[0]);

    f32x8 acc[TPW];
#pragma unroll
    for (int t = 0; t < TPW; ++t)
        acc[t] = (f32x8){0.f, 0.f, 0.f, 0.f, 0.f, 0.f, 0.f, 0.f};

    for (int term = 0; term <= RR; ++term) {
        // ---- async-stage B (swizzled 128 x OUTC bf16) via TDM, wave 0 only ----
        if (wave == 0) {
            const __bf16* Bsrc = (term < RR) ? (wrel + (size_t)term * DD * OUTC) : root;
            tdm_load_1d(sB_off, Bsrc, (DD * OUTC * 2) / 4);
        }
        // ---- stage A tile (64 x 128 bf16, row-major) ----
        if (term < RR) {
            for (int e = tid; e < 64 * DD; e += 256) {
                int row = e >> 7, col = e & 127;
                int n = node0 + row;
                float v = 0.f;
                if (n < n_nodes) {
                    float ct = cnt[(size_t)term * n_nodes + n];
                    float sc = 1.0f / fmaxf(ct, 1.0f);
                    v = agg[((size_t)term * n_nodes + n) * DD + col] * sc;
                }
                sA[e] = (__bf16)v;
            }
        } else {
            for (int e = tid; e < 64 * DD; e += 256) {
                int row = e >> 7, col = e & 127;
                int n = node0 + row;
                sA[e] = (n < n_nodes) ? xin[(size_t)n * DD + col] : (__bf16)0.f;
            }
        }
        if (wave == 0) __builtin_amdgcn_s_wait_tensorcnt(0);
        __syncthreads();

        // ---- accumulate WMMA tiles ----
#pragma unroll
        for (int t = 0; t < TPW; ++t) {
            int tile = wave * TPW + t;
            int rt = tile / NCT, ct = tile % NCT;
#pragma unroll
            for (int kk = 0; kk < 4; ++kk) {
                bf16x16 a = load_a_frag(&sA[(rt * 16) * DD + kk * 32], DD, lane);
                bf16x16 b = load_b_frag_sw(&sB[(size_t)(kk * NCT + ct) * 512], lane);
                acc[t] = wmma_bf16(a, b, acc[t]);
            }
        }
        __syncthreads();
    }

    // ---- epilogue: bias + activation + store ----
    int n = lane & 15, half = lane >> 4;
#pragma unroll
    for (int t = 0; t < TPW; ++t) {
        int tile = wave * TPW + t;
        int rt = tile / NCT, ct = tile % NCT;
        int col = ct * 16 + n;
        float bs = bias[col];
#pragma unroll
        for (int i = 0; i < 8; ++i) {
            int m = i + half * 8;
            int node = node0 + rt * 16 + m;
            if (node >= n_nodes) continue;
            float v = acc[t][i] + bs;
            if (MODE == 0) {
                v = fmaxf(v, 0.f);
                ((__bf16*)out)[(size_t)node * OUTC + col] = (__bf16)v;
            } else {
                v = 1.0f / (1.0f + __expf(-v));
                ((float*)out)[(size_t)node * OUTC + col] = v;
            }
        }
    }
}

// ---------------------------------------------------------------------------
// Host-side launcher
// ---------------------------------------------------------------------------
extern "C" void kernel_launch(void* const* d_in, const int* in_sizes, int n_in,
                              void* d_out, int out_size, void* d_ws, size_t ws_size,
                              hipStream_t stream) {
    const int*   edge_index = (const int*)d_in[0];
    const int*   edge_type  = (const int*)d_in[1];
    const float* emb        = (const float*)d_in[2];
    const float* in_proj_w  = (const float*)d_in[3];
    const float* in_proj_b  = (const float*)d_in[4];
    const float* out_proj_w = (const float*)d_in[5];
    const float* out_proj_b = (const float*)d_in[6];
    const float* w1         = (const float*)d_in[7];
    const float* root1      = (const float*)d_in[8];
    const float* b1         = (const float*)d_in[9];
    const float* w2         = (const float*)d_in[10];
    const float* root2      = (const float*)d_in[11];
    const float* b2         = (const float*)d_in[12];

    const int n_edges = in_sizes[1];
    const int n_nodes = in_sizes[2] / (SS * DD);

    // ---- workspace carve-up ----
    char* p = (char*)d_ws;
    size_t off = 0;
    auto take = [&](size_t bytes) -> char* {
        char* r = p + off;
        off += (bytes + 255) & ~(size_t)255;
        return r;
    };
    __bf16* wiS   = (__bf16*)take((size_t)DD * 3 * DD * 2);
    __bf16* woS   = (__bf16*)take((size_t)DD * DD * 2);
    __bf16* w1S   = (__bf16*)take((size_t)RR * DD * HIDC * 2);
    __bf16* r1S   = (__bf16*)take((size_t)DD * HIDC * 2);
    __bf16* w2S   = (__bf16*)take((size_t)RR * HIDC * LL * 2);
    __bf16* r2S   = (__bf16*)take((size_t)HIDC * LL * 2);
    __bf16* x0    = (__bf16*)take((size_t)n_nodes * DD * 2);
    __bf16* x1    = (__bf16*)take((size_t)n_nodes * DD * 2);
    float*  cnt   = (float*)take((size_t)RR * n_nodes * 4);
    float*  agg   = (float*)take((size_t)RR * n_nodes * DD * 4);

    const size_t agg_bytes = (size_t)RR * n_nodes * DD * 4;
    const size_t cnt_bytes = (size_t)RR * n_nodes * 4;

    // 0: weights -> bf16 (pre-swizzled B blocks)
    {
        int total = 3 * DD * DD + DD * DD + RR * DD * HIDC + DD * HIDC +
                    RR * HIDC * LL + HIDC * LL;
        convert_weights_kernel<<<(total + 255) / 256, 256, 0, stream>>>(
            in_proj_w, out_proj_w, w1, root1, w2, root2,
            wiS, woS, w1S, r1S, w2S, r2S);
    }

    // 1: MHA -> x0
    mha_kernel<<<(n_nodes + 15) / 16, 256, 0, stream>>>(
        emb, wiS, in_proj_b, woS, out_proj_b, x0, n_nodes);

    // 2: counts (shared by both layers)
    (void)hipMemsetAsync(cnt, 0, cnt_bytes, stream);
    count_kernel<<<(n_edges + 255) / 256, 256, 0, stream>>>(
        edge_index, edge_type, cnt, n_edges, n_nodes);

    // 3: layer-1 aggregation + GEMM (ReLU -> x1 bf16)
    (void)hipMemsetAsync(agg, 0, agg_bytes, stream);
    {
        long long thr = (long long)n_edges * 32;
        aggregate_kernel<<<(unsigned)((thr + 255) / 256), 256, 0, stream>>>(
            edge_index, edge_type, x0, agg, n_edges, n_nodes);
    }
    rgcn_kernel<HIDC, 0><<<(n_nodes + 63) / 64, 256, 0, stream>>>(
        agg, cnt, x0, w1S, r1S, b1, (void*)x1, n_nodes);

    // 4: layer-2 aggregation + GEMM (sigmoid -> d_out f32)
    (void)hipMemsetAsync(agg, 0, agg_bytes, stream);
    {
        long long thr = (long long)n_edges * 32;
        aggregate_kernel<<<(unsigned)((thr + 255) / 256), 256, 0, stream>>>(
            edge_index, edge_type, x1, agg, n_edges, n_nodes);
    }
    rgcn_kernel<LL, 1><<<(n_nodes + 63) / 64, 256, 0, stream>>>(
        agg, cnt, x1, w2S, r2S, b2, d_out, n_nodes);
}
